// Block_86028194939235
// MI455X (gfx1250) — matrix operand order repaired and checked by reading, not compiled, not run on hardware
//
#include <hip/hip_runtime.h>
#include <math.h>

#define B_ 2
#define T_ 1024
#define C_ 1024
#define H_ 16
#define D_ 64

typedef _Float16 half_t;
typedef __attribute__((ext_vector_type(16))) _Float16 v16h;
typedef __attribute__((ext_vector_type(8)))  _Float16 v8h;
typedef __attribute__((ext_vector_type(8)))  float    v8f;

// ---------------- WMMA fragment helpers (gfx1250 wave32 layouts) ----------------

// A-matrix 16x32 f16: lanes 0-15 -> M=0..15, elems 0-7 -> K=h*8+0..7, elems 8-15 -> K=16+h*8+0..7
static __device__ __forceinline__ v16h load_frag_a(const half_t* base, int ld, int row0, int k0, int lane) {
  const int lr = lane & 15, hf = lane >> 4;
  const half_t* p = base + (size_t)(row0 + lr) * (size_t)ld + k0 + hf * 8;
  v8h lo = *(const v8h*)(p);
  v8h hi = *(const v8h*)(p + 16);
  v16h r;
#pragma unroll
  for (int e = 0; e < 8; ++e) { r[e] = lo[e]; r[8 + e] = hi[e]; }
  return r;
}

// B-matrix 32x16 f16: lane -> N=lane%16, elems 0-7 -> K=h*8+0..7, elems 8-15 -> K=16+h*8+0..7
static __device__ __forceinline__ v16h load_frag_b(const half_t* base, int ld, int col0, int k0, int lane) {
  const int lr = lane & 15, hf = lane >> 4;
  const half_t* p = base + (size_t)(col0 + lr) * (size_t)ld + k0 + hf * 8;
  v8h lo = *(const v8h*)(p);
  v8h hi = *(const v8h*)(p + 16);
  v16h r;
#pragma unroll
  for (int e = 0; e < 8; ++e) { r[e] = lo[e]; r[8 + e] = hi[e]; }
  return r;
}

static __device__ __forceinline__ v8f wmma32(v16h a, v16h b, v8f c) {
  return __builtin_amdgcn_wmma_f32_16x16x32_f16(false, a, false, b, (short)0, c, false, false);
}

static __device__ __forceinline__ v8f zero8() {
  v8f z = {0.f, 0.f, 0.f, 0.f, 0.f, 0.f, 0.f, 0.f};
  return z;
}

// ---------------- generic GEMM:  C[M,N] = A[M,K] @ W[N,K]^T ----------------
// block = 256 threads = 8 waves (2 M x 4 N); wave tile 32x64; block tile 64x256.
template <int OUT_HALF, int COL_SCALE>
__global__ __launch_bounds__(256) void gemm_f16_wmma(const half_t* __restrict__ A,
                                                     const half_t* __restrict__ Bm,
                                                     void* __restrict__ Cout,
                                                     const float* __restrict__ colscale,
                                                     float scalemul,
                                                     int M, int N, int K,
                                                     int lda, int ldb, int ldc) {
  const int lane = threadIdx.x & 31;
  const int wave = threadIdx.x >> 5;
  const int wm = wave & 1, wn = wave >> 1;      // 2 x 4 waves
  const int m0 = blockIdx.y * 64 + wm * 32;     // 32 rows per wave
  const int n0 = blockIdx.x * 256 + wn * 64;    // 64 cols per wave
  (void)M; (void)N;

  v8f acc[2][4];
#pragma unroll
  for (int t = 0; t < 2; ++t)
#pragma unroll
    for (int j = 0; j < 4; ++j) acc[t][j] = zero8();

  for (int k0 = 0; k0 < K; k0 += 32) {
    if (k0 + 64 < K) {
      __builtin_prefetch(A + (size_t)(m0 + (lane & 15)) * (size_t)lda + k0 + 64, 0, 0);
      __builtin_prefetch(Bm + (size_t)(n0 + lane) * (size_t)ldb + k0 + 64, 0, 0);
    }
    const v16h a0 = load_frag_a(A, lda, m0,      k0, lane);
    const v16h a1 = load_frag_a(A, lda, m0 + 16, k0, lane);
#pragma unroll
    for (int j = 0; j < 4; ++j) {
      const v16h bb = load_frag_b(Bm, ldb, n0 + j * 16, k0, lane);
      acc[0][j] = wmma32(a0, bb, acc[0][j]);
      acc[1][j] = wmma32(a1, bb, acc[1][j]);
    }
  }

  const int lr = lane & 15, hf = lane >> 4;
#pragma unroll
  for (int t = 0; t < 2; ++t)
#pragma unroll
    for (int j = 0; j < 4; ++j) {
      const int n = n0 + j * 16 + lr;
      float cs = 1.f;
      if (COL_SCALE) cs = colscale[n] * scalemul;
#pragma unroll
      for (int i = 0; i < 8; ++i) {
        const int m = m0 + t * 16 + 8 * hf + i;
        const float val = acc[t][j][i] * cs;
        if (OUT_HALF) ((half_t*)Cout)[(size_t)m * (size_t)ldc + n] = (half_t)val;
        else          ((float*) Cout)[(size_t)m * (size_t)ldc + n] = val;
      }
    }
}

// ---------------- fp32 -> fp16 cast ----------------
__global__ __launch_bounds__(256) void cvt_f32_f16(const float* __restrict__ in,
                                                   half_t* __restrict__ out, size_t n) {
  for (size_t i = (size_t)blockIdx.x * blockDim.x + threadIdx.x; i < n;
       i += (size_t)gridDim.x * blockDim.x)
    out[i] = (half_t)in[i];
}

// ---------------- V transpose: (B,T,C) f32 -> (B,H,D,T) f16 ----------------
__global__ __launch_bounds__(256) void v_transpose(const float* __restrict__ vsrc,
                                                   half_t* __restrict__ vT) {
  const size_t n = (size_t)B_ * T_ * C_;
  for (size_t i = (size_t)blockIdx.x * blockDim.x + threadIdx.x; i < n;
       i += (size_t)gridDim.x * blockDim.x) {
    const int c = (int)(i % C_);
    const int t = (int)((i / C_) % T_);
    const int b = (int)(i / ((size_t)T_ * C_));
    const int h = c >> 6, d = c & 63;
    vT[(((size_t)b * H_ + h) * D_ + d) * T_ + t] = (half_t)vsrc[i];
  }
}

// ---------------- RoPE + justnorm + sqk scale (one wave per (b,t,head)) ----------------
__global__ __launch_bounds__(256) void rope_norm_qk(const float* __restrict__ src,
                                                    half_t* __restrict__ dst,
                                                    const float* __restrict__ sqk) {
  const int gid = blockIdx.x * blockDim.x + threadIdx.x;
  const int lane = threadIdx.x & 31;       // pair index i = lane, 0..31
  const int w = gid >> 5;                  // (b*T + t)*H + h
  if (w >= B_ * T_ * H_) return;
  const int hh = w % H_;
  const int bt = w / H_;
  const int t = bt % T_;

  const float* p = src + (size_t)bt * C_ + hh * D_;
  const float x0 = p[2 * lane];
  const float x1 = p[2 * lane + 1];

  // emb(t, j): u=j/2, div=10000^(-u/32); even j -> sin, odd j -> cos
  const float lc = logf(10000.f) / 64.f;
  const int ja = 32 + lane, jb = lane;
  const int ua = ja >> 1, ub = jb >> 1;
  const float anga = (float)t * __expf(-(float)(2 * ua) * lc);
  const float angb = (float)t * __expf(-(float)(2 * ub) * lc);
  const float ea = (ja & 1) ? __cosf(anga) : __sinf(anga);
  const float eb = (jb & 1) ? __cosf(angb) : __sinf(angb);

  float o0 = -x1 * ea;  // out[2i]   = -x[2i+1] * emb[t,32+i]
  float o1 =  x0 * eb;  // out[2i+1] =  x[2i]   * emb[t,i]

  float ss = o0 * o0 + o1 * o1;
#pragma unroll
  for (int off = 16; off; off >>= 1) ss += __shfl_xor(ss, off, 32);
  const float inv = rsqrtf(ss);

  const float s0 = sqk[hh * D_ + 2 * lane]     * 32.f;  // 1/BASE_SCALE
  const float s1 = sqk[hh * D_ + 2 * lane + 1] * 32.f;
  dst[(size_t)bt * C_ + hh * D_ + 2 * lane]     = (half_t)(o0 * inv * s0);
  dst[(size_t)bt * C_ + hh * D_ + 2 * lane + 1] = (half_t)(o1 * inv * s1);
}

// ---------------- gated causal attention: one wave per (b, h, 16-query tile) ----------------
// y = normalize( sum_k exp(s-m)*sigmoid(stp*(s-thr)) * v ),  s = (q.k)*sqrt(D)
// Causal split: tiles with kt < q0 are fully valid (no predicates); only the
// diagonal tile (kt == q0) is masked.
__global__ __launch_bounds__(32) void attn_wmma(const half_t* __restrict__ q16,
                                                const half_t* __restrict__ k16,
                                                const half_t* __restrict__ vT16,
                                                half_t* __restrict__ y16,
                                                const float* __restrict__ thr_c,
                                                const float* __restrict__ stp) {
  __shared__ __attribute__((aligned(32))) half_t Pl[16 * 32];

  const int lane = threadIdx.x;
  const int q0 = blockIdx.x * 16;
  const int hh = blockIdx.y;
  const int b  = blockIdx.z;
  const int lr = lane & 15, hf = lane >> 4;

  const half_t* qb  = q16  + (size_t)b * T_ * C_ + hh * D_;
  const half_t* kb  = k16  + (size_t)b * T_ * C_ + hh * D_;
  const half_t* vTb = vT16 + ((size_t)b * H_ + hh) * D_ * T_;

  const v16h aq0 = load_frag_a(qb, C_, q0, 0,  lane);
  const v16h aq1 = load_frag_a(qb, C_, q0, 32, lane);
  const float thr = thr_c[hh], st = stp[hh];
  const int nk = q0 + 16;  // causal: k positions < q0+16

  // ---- pass 1: per-row max of scaled scores ----
  v8f mloc;
#pragma unroll
  for (int i = 0; i < 8; ++i) mloc[i] = -3.0e38f;
  for (int kt = 0; kt < q0; kt += 16) {        // bulk: fully valid, no mask
    v8f s = zero8();
    s = wmma32(aq0, load_frag_b(kb, C_, kt, 0,  lane), s);
    s = wmma32(aq1, load_frag_b(kb, C_, kt, 32, lane), s);
#pragma unroll
    for (int i = 0; i < 8; ++i) mloc[i] = fmaxf(mloc[i], s[i] * 8.0f);
  }
  {                                            // diagonal tile: masked
    v8f s = zero8();
    s = wmma32(aq0, load_frag_b(kb, C_, q0, 0,  lane), s);
    s = wmma32(aq1, load_frag_b(kb, C_, q0, 32, lane), s);
    const int kpos = q0 + lr;
#pragma unroll
    for (int i = 0; i < 8; ++i) {
      const int row = q0 + 8 * hf + i;
      if (kpos <= row) mloc[i] = fmaxf(mloc[i], s[i] * 8.0f);
    }
  }
#pragma unroll
  for (int off = 1; off < 16; off <<= 1)
#pragma unroll
    for (int i = 0; i < 8; ++i) mloc[i] = fmaxf(mloc[i], __shfl_xor(mloc[i], off, 32));

  // ---- pass 2: exp * gate, row sums, and P @ V via WMMA (K chunks of 32) ----
  v8f ssum = zero8();
  v8f yacc[4] = {zero8(), zero8(), zero8(), zero8()};

  for (int k0 = 0; k0 < nk; k0 += 32) {
#pragma unroll
    for (int t2 = 0; t2 < 2; ++t2) {
      const int kk = k0 + t2 * 16;
      if (kk < q0) {                           // fully-valid tile: no predicates
        v8f s = zero8();
        s = wmma32(aq0, load_frag_b(kb, C_, kk, 0,  lane), s);
        s = wmma32(aq1, load_frag_b(kb, C_, kk, 32, lane), s);
#pragma unroll
        for (int i = 0; i < 8; ++i) {
          const float sv = s[i] * 8.0f;
          const float g = 1.f / (1.f + __expf(-st * (sv - thr)));
          const float p = __expf(sv - mloc[i]) * g;
          ssum[i] += p;
          Pl[(8 * hf + i) * 32 + t2 * 16 + lr] = (half_t)p;
        }
      } else if (kk == q0) {                   // diagonal tile: masked
        v8f s = zero8();
        s = wmma32(aq0, load_frag_b(kb, C_, kk, 0,  lane), s);
        s = wmma32(aq1, load_frag_b(kb, C_, kk, 32, lane), s);
        const int kpos = kk + lr;
#pragma unroll
        for (int i = 0; i < 8; ++i) {
          const int row = q0 + 8 * hf + i;
          float p = 0.f;
          if (kpos <= row) {
            const float sv = s[i] * 8.0f;
            const float g = 1.f / (1.f + __expf(-st * (sv - thr)));
            p = __expf(sv - mloc[i]) * g;
            ssum[i] += p;
          }
          Pl[(8 * hf + i) * 32 + t2 * 16 + lr] = (half_t)p;
        }
      } else {                                 // beyond causal frontier: zeros
#pragma unroll
        for (int i = 0; i < 8; ++i)
          Pl[(8 * hf + i) * 32 + t2 * 16 + lr] = (half_t)0.f;
      }
    }
    __syncthreads();  // single-wave WG: orders LDS write -> read
    const v16h pa = load_frag_a((const half_t*)Pl, 32, 0, 0, lane);
    __syncthreads();
#pragma unroll
    for (int j = 0; j < 4; ++j)
      yacc[j] = wmma32(pa, load_frag_b(vTb, T_, j * 16, k0, lane), yacc[j]);
  }

#pragma unroll
  for (int off = 1; off < 16; off <<= 1)
#pragma unroll
    for (int i = 0; i < 8; ++i) ssum[i] += __shfl_xor(ssum[i], off, 32);

#pragma unroll
  for (int j = 0; j < 4; ++j)
#pragma unroll
    for (int i = 0; i < 8; ++i) {
      const int row = q0 + 8 * hf + i;
      y16[((size_t)b * T_ + row) * C_ + hh * D_ + j * 16 + lr] =
          (half_t)(yacc[j][i] / ssum[i]);
    }
}

// ---------------- nGPT residual update: out = justnorm(A + lr*(Bn - A)) ----------------
static __device__ __forceinline__ float block_reduce_sum(float v, float* sbuf) {
#pragma unroll
  for (int off = 16; off; off >>= 1) v += __shfl_xor(v, off, 32);
  __syncthreads();
  if ((threadIdx.x & 31) == 0) sbuf[threadIdx.x >> 5] = v;
  __syncthreads();
  float r = 0.f;
#pragma unroll
  for (int j = 0; j < 8; ++j) r += sbuf[j];
  return r;
}

__global__ __launch_bounds__(256) void resnorm(const float* __restrict__ base,
                                               const float* __restrict__ delta,
                                               const float* __restrict__ alpha,
                                               float alpha_mul,
                                               float* __restrict__ out_f32,
                                               half_t* __restrict__ out_f16) {
  __shared__ float sbuf[8];
  const size_t row = blockIdx.x;
  const float* hb = base  + row * C_;
  const float* db = delta + row * C_;

  float a4[4], d4[4];
  float ssA = 0.f, ssB = 0.f;
#pragma unroll
  for (int i = 0; i < 4; ++i) {
    const int c = threadIdx.x + i * 256;
    a4[i] = hb[c]; d4[i] = db[c];
    ssA += a4[i] * a4[i]; ssB += d4[i] * d4[i];
  }
  ssA = block_reduce_sum(ssA, sbuf);
  ssB = block_reduce_sum(ssB, sbuf);
  const float invA = rsqrtf(ssA), invB = rsqrtf(ssB);

  float t4[4], ssT = 0.f;
#pragma unroll
  for (int i = 0; i < 4; ++i) {
    const int c = threadIdx.x + i * 256;
    const float lr = fabsf(alpha[c] * alpha_mul);
    const float A = a4[i] * invA;
    t4[i] = A + lr * (d4[i] * invB - A);
    ssT += t4[i] * t4[i];
  }
  ssT = block_reduce_sum(ssT, sbuf);
  const float invT = rsqrtf(ssT);

#pragma unroll
  for (int i = 0; i < 4; ++i) {
    const int c = threadIdx.x + i * 256;
    const float o = t4[i] * invT;
    out_f32[row * C_ + c] = o;
    if (out_f16) out_f16[row * C_ + c] = (half_t)o;
  }
}

// ---------------- SwiGLU: x[m,j] = u * v * sigmoid(v), u=uv[m,j], v=uv[m,4096+j] ----------------
__global__ __launch_bounds__(256) void swiglu(const half_t* __restrict__ uv,
                                              half_t* __restrict__ x, size_t n) {
  for (size_t i = (size_t)blockIdx.x * blockDim.x + threadIdx.x; i < n;
       i += (size_t)gridDim.x * blockDim.x) {
    const size_t m = i / (4 * C_);
    const size_t j = i % (4 * C_);
    const float u  = (float)uv[m * (8 * C_) + j];
    const float vv = (float)uv[m * (8 * C_) + 4 * C_ + j];
    x[i] = (half_t)(u * (vv / (1.f + __expf(-vv))));
  }
}

// ---------------- host-side orchestration ----------------
extern "C" void kernel_launch(void* const* d_in, const int* in_sizes, int n_in,
                              void* d_out, int out_size, void* d_ws, size_t ws_size,
                              hipStream_t stream) {
  (void)in_sizes; (void)n_in; (void)out_size; (void)ws_size;

  const float* h        = (const float*)d_in[0];
  const float* Wq       = (const float*)d_in[1];
  const float* Wk       = (const float*)d_in[2];
  const float* Wv       = (const float*)d_in[3];
  const float* Wo       = (const float*)d_in[4];
  const float* Wfc      = (const float*)d_in[5];
  const float* Wproj    = (const float*)d_in[6];
  const float* sqk      = (const float*)d_in[7];
  const float* suv      = (const float*)d_in[8];
  const float* a_attn   = (const float*)d_in[9];
  const float* a_mlp    = (const float*)d_in[10];
  const float* thr_c    = (const float*)d_in[11];
  const float* stp      = (const float*)d_in[12];
  float* out = (float*)d_out;

  const size_t BT = (size_t)B_ * T_;           // 2048
  char* ws = (char*)d_ws;
  size_t off = 0;
  auto take = [&](size_t bytes) { char* p = ws + off; off += (bytes + 255) & ~(size_t)255; return p; };

  half_t* h16     = (half_t*)take(BT * C_ * 2);
  half_t* Wq16    = (half_t*)take((size_t)C_ * C_ * 2);
  half_t* Wk16    = (half_t*)take((size_t)C_ * C_ * 2);
  half_t* Wv16    = (half_t*)take((size_t)C_ * C_ * 2);
  half_t* Wo16    = (half_t*)take((size_t)C_ * C_ * 2);
  half_t* Wfc16   = (half_t*)take((size_t)8 * C_ * C_ * 2);
  half_t* Wproj16 = (half_t*)take((size_t)4 * C_ * C_ * 2);
  half_t* q16     = (half_t*)take(BT * C_ * 2);
  half_t* k16     = (half_t*)take(BT * C_ * 2);
  half_t* vT16    = (half_t*)take(BT * C_ * 2);
  half_t* y16     = (half_t*)take(BT * C_ * 2);
  float*  h_att   = (float*) take(BT * C_ * 4);
  float*  h1      = (float*) take(BT * C_ * 4);
  half_t* h1_16   = (half_t*)take(BT * C_ * 2);
  float*  h_mlp   = (float*) take(BT * C_ * 4);
  // union region: {qf32,kf32,vf32} early  |  later {uv16, x16}
  char* U = take((size_t)48 << 20);
  float*  qf32 = (float*)U;
  float*  kf32 = (float*)(U + ((size_t)8 << 20));
  float*  vf32 = (float*)(U + ((size_t)16 << 20));
  half_t* uv16 = (half_t*)U;                               // 32 MB
  half_t* x16  = (half_t*)(U + ((size_t)32 << 20));        // 16 MB

  // 1) casts to f16
  cvt_f32_f16<<<2048, 256, 0, stream>>>(h,     h16,     BT * C_);
  cvt_f32_f16<<<1024, 256, 0, stream>>>(Wq,    Wq16,    (size_t)C_ * C_);
  cvt_f32_f16<<<1024, 256, 0, stream>>>(Wk,    Wk16,    (size_t)C_ * C_);
  cvt_f32_f16<<<1024, 256, 0, stream>>>(Wv,    Wv16,    (size_t)C_ * C_);
  cvt_f32_f16<<<1024, 256, 0, stream>>>(Wo,    Wo16,    (size_t)C_ * C_);
  cvt_f32_f16<<<4096, 256, 0, stream>>>(Wfc,   Wfc16,   (size_t)8 * C_ * C_);
  cvt_f32_f16<<<2048, 256, 0, stream>>>(Wproj, Wproj16, (size_t)4 * C_ * C_);

  // 2) QKV projections (f16 WMMA, f32 out)
  dim3 gq(C_ / 256, BT / 64);
  gemm_f16_wmma<0, 0><<<gq, 256, 0, stream>>>(h16, Wq16, qf32, nullptr, 1.f, (int)BT, C_, C_, C_, C_, C_);
  gemm_f16_wmma<0, 0><<<gq, 256, 0, stream>>>(h16, Wk16, kf32, nullptr, 1.f, (int)BT, C_, C_, C_, C_, C_);
  gemm_f16_wmma<0, 0><<<gq, 256, 0, stream>>>(h16, Wv16, vf32, nullptr, 1.f, (int)BT, C_, C_, C_, C_, C_);

  // 3) RoPE + normalize + sqk scale; transpose V
  rope_norm_qk<<<(B_ * T_ * H_ * 32) / 256, 256, 0, stream>>>(qf32, q16, sqk);
  rope_norm_qk<<<(B_ * T_ * H_ * 32) / 256, 256, 0, stream>>>(kf32, k16, sqk);
  v_transpose<<<2048, 256, 0, stream>>>(vf32, vT16);

  // 4) gated causal attention
  attn_wmma<<<dim3(T_ / 16, H_, B_), 32, 0, stream>>>(q16, k16, vT16, y16, thr_c, stp);

  // 5) output projection + attention residual-norm
  gemm_f16_wmma<0, 0><<<gq, 256, 0, stream>>>(y16, Wo16, h_att, nullptr, 1.f, (int)BT, C_, C_, C_, C_, C_);
  resnorm<<<(int)BT, 256, 0, stream>>>(h, h_att, a_attn, 1.6f, h1, h1_16);

  // 6) MLP: fc (with suv*sqrt(C) column scale, f16 out), SwiGLU, proj
  gemm_f16_wmma<1, 1><<<dim3(8 * C_ / 256, BT / 64), 256, 0, stream>>>(
      h1_16, Wfc16, uv16, suv, 32.f, (int)BT, 8 * C_, C_, C_, C_, 8 * C_);
  swiglu<<<4096, 256, 0, stream>>>(uv16, x16, BT * (size_t)(4 * C_));
  gemm_f16_wmma<0, 0><<<dim3(C_ / 256, BT / 64), 256, 0, stream>>>(
      x16, Wproj16, h_mlp, nullptr, 1.f, (int)BT, C_, 4 * C_, 4 * C_, 4 * C_, C_);

  // 7) MLP residual-norm -> final output
  resnorm<<<(int)BT, 256, 0, stream>>>(h1, h_mlp, a_mlp, 1.6f, out, nullptr);
}
